// WindowBasedFrequencyChannelAttention_76149770158724
// MI455X (gfx1250) — matrix operand order repaired and checked by reading, not compiled, not run on hardware
//
#include <hip/hip_runtime.h>
#include <math.h>

// MI455X / gfx1250 fused Window-FFT-Channel-Attention.
// One workgroup (8 wave32) == one 16x16 window x 96 channels.
// FFT / 1x1 conv / SE / IFFT as v_wmma_f32_16x16x32_f16 with full-K stacking;
// weights staged via TDM tensor_load_to_lds overlapped with the FFT stage.

typedef __attribute__((ext_vector_type(16))) _Float16 v16h;
typedef __attribute__((ext_vector_type(8)))  _Float16 v8h;
typedef __attribute__((ext_vector_type(8)))  float    v8f;

#if defined(__has_builtin)
#if __has_builtin(__builtin_amdgcn_tensor_load_to_lds)
#define USE_TDM 1
#endif
#endif
#ifndef USE_TDM
#define USE_TDM 0
#endif

namespace {

constexpr int WSZ   = 16;
constexpr int CCH   = 96;
constexpr int CRED  = 32;
constexpr int PWIN  = WSZ * WSZ;     // 256
constexpr int BLK   = 256;           // 8 wave32
constexpr int NWAVE = BLK / 32;
constexpr int NWIN  = 8 * 16 * 16;   // 2048 windows

// dynamic-LDS layout (offsets in halves)
constexpr int OFF_RE  = 0;                      // spectrum real, [q][c] q=u*16+v
constexpr int OFF_IM  = OFF_RE  + CCH * PWIN;   // spectrum imag, [q][c]
constexpr int OFF_CR  = OFF_IM  + CCH * PWIN;   // conv real, [o][v*16+u]  (also TDM stage)
constexpr int OFF_CI  = OFF_CR  + CCH * PWIN;   // conv imag, [o][v*16+u]
constexpr int OFF_WR  = OFF_CI  + CCH * PWIN;   // w_real f16 [o][c]
constexpr int OFF_WI  = OFF_WR  + CCH * CCH;    // w_imag f16
constexpr int OFF_FR  = OFF_WI  + CCH * CCH;    // DFT cos [u][r]
constexpr int OFF_FI  = OFF_FR  + 256;          // DFT sin
constexpr int OFF_SCR = OFF_FI  + 256;          // per-wave 16x32 stacked scratch
constexpr int HALVES  = OFF_SCR + NWAVE * 512;
constexpr int NF32    = 2 * CCH + 2 * CRED + 2 * CCH + 2 * CCH;  // pool,hid,scale,bias
constexpr size_t SMEM_BYTES = (size_t)HALVES * 2 + (size_t)NF32 * 4;   // ~240 KB (245248 B)

__device__ __forceinline__ v8f wmma_f16(v16h a, v16h b, v8f c) {
  return __builtin_amdgcn_wmma_f32_16x16x32_f16(false, a, false, b, (short)0, c,
                                                false, false);
}

__device__ __forceinline__ v16h frag_from2(v8h lo, v8h hi8) {
  v16h r;
#pragma unroll
  for (int e = 0; e < 8; ++e) { r[e] = lo[e]; r[e + 8] = hi8[e]; }
  return r;
}

// A fragment (16xK=32), row-major, stride ld halves (ld%8==0, 16B-aligned base).
// k(e,lane) = e + 16*(lane>=16)  -> two contiguous b128 loads
__device__ __forceinline__ v16h frag_a_k32v(const _Float16* m, int ld) {
  const int lane = threadIdx.x & 31;
  const _Float16* p = m + (lane & 15) * ld + (lane >> 4) * 16;
  return frag_from2(*(const v8h*)p, *(const v8h*)(p + 8));
}

// B fragment (K=32 x 16) from transposed storage: element (k,n) at m[n*ldn+k].
// k(e,lane) = e + 16*(lane>=16) -> 16 contiguous halves per lane
__device__ __forceinline__ v16h frag_b_k32t(const _Float16* m, int ldn) {
  const int lane = threadIdx.x & 31;
  const _Float16* p = m + (lane & 15) * ldn + (lane >> 4) * 16;
  return frag_from2(*(const v8h*)p, *(const v8h*)(p + 8));
}

// A fragment for stacked [Lo | sHi*Hi] (two 16x16 row-major ld16 matrices).
__device__ __forceinline__ v16h frag_a_pair16(const _Float16* lo, const _Float16* hiM,
                                              float sHi) {
  const int lane = threadIdx.x & 31;
  const int r = lane & 15;
  const _Float16* src = (lane >> 4) ? hiM : lo;
  const float s = (lane >> 4) ? sHi : 1.0f;
  v16h a;
#pragma unroll
  for (int e = 0; e < 16; ++e) a[e] = (_Float16)((float)src[r * 16 + e] * s);
  return a;
}

// B fragment for stacked [sLo*Lo ; sHi*Hi] (two 16x16 row-major ld16 matrices).
__device__ __forceinline__ v16h frag_b_pair16(const _Float16* lo, const _Float16* hiM,
                                              float sLo, float sHi) {
  const int lane = threadIdx.x & 31;
  const int n = lane & 15;
  const _Float16* src = (lane >> 4) ? hiM : lo;
  const float s = (lane >> 4) ? sHi : sLo;
  v16h b;
#pragma unroll
  for (int e = 0; e < 16; ++e) b[e] = (_Float16)((float)src[e * 16 + n] * s);
  return b;
}

// B fragment [sLo*ZLo ; sHi*ZHi], each Z stored per-channel as [n][k] contiguous.
__device__ __forceinline__ v16h frag_b_zz(const _Float16* loM, const _Float16* hiM,
                                          _Float16 sLo, _Float16 sHi) {
  const int lane = threadIdx.x & 31;
  const _Float16* src = (lane >> 4) ? hiM : loM;
  const _Float16 s = (lane >> 4) ? sHi : sLo;
  const v8h* p = (const v8h*)(src + (lane & 15) * 16);
  return frag_from2(p[0] * s, p[1] * s);
}

// Padded A fragment of one 16x16 matrix (K 16..31 zero).
__device__ __forceinline__ v16h frag_a_k16(const _Float16* m16, float sgn) {
  const int lane = threadIdx.x & 31;
  const int r = lane & 15, kh = (lane >> 4) * 8;
  v16h a;
#pragma unroll
  for (int e = 0; e < 8; ++e) a[e] = (_Float16)((float)m16[r * 16 + kh + e] * sgn);
#pragma unroll
  for (int e = 8; e < 16; ++e) a[e] = (_Float16)0.0f;
  return a;
}

// Store f32 D fragment as f16: addr = base + n*cs + (m0+r)*rs
__device__ __forceinline__ void store_d_rc(_Float16* base, int rs, int cs, v8f d) {
  const int lane = threadIdx.x & 31;
  _Float16* p = base + (lane & 15) * cs + ((lane >> 4) * 8) * rs;
#pragma unroll
  for (int r = 0; r < 8; ++r) p[r * rs] = (_Float16)d[r];
}

#if USE_TDM
typedef __attribute__((ext_vector_type(4))) unsigned int tdm_v4u;
typedef __attribute__((ext_vector_type(8))) int tdm_v8i;
typedef __attribute__((ext_vector_type(4))) int tdm_v4i;

// 1-D TDM load: nelem fp32 elements, contiguous, global -> LDS byte offset.
__device__ __forceinline__ void tdm_load_linear(unsigned lds_byte, const float* src,
                                                unsigned nelem) {
  const unsigned long long ga = (unsigned long long)(uintptr_t)src;
  tdm_v4u g0;
  g0[0] = 1u;                                   // count=1, user descriptor
  g0[1] = lds_byte;                             // lds_addr
  g0[2] = (unsigned)(ga & 0xffffffffu);         // global_addr[31:0]
  g0[3] = (unsigned)((ga >> 32) & 0x01ffffffu)  // global_addr[56:32]
        | (2u << 30);                           // type = 2 ("image")
  tdm_v8i g1;
  g1[0] = (int)(2u << 16);                      // data_size = 4B, wg_mask = 0
  g1[1] = (int)((nelem & 0xffffu) << 16);       // tensor_dim0[15:0]
  g1[2] = (int)(((nelem >> 16) & 0xffffu) | (1u << 16));  // dim0[31:16], tensor_dim1=1
  g1[3] = (int)((nelem & 0xffffu) << 16);       // tile_dim0 = nelem (16-bit)
  g1[4] = 1;                                    // tile_dim1 = 1, tile_dim2 = 0
  g1[5] = (int)nelem;                           // tensor_dim0_stride
  g1[6] = 0;
  g1[7] = 0;
  const tdm_v4i z4 = {0, 0, 0, 0};
  const tdm_v8i z8 = {0, 0, 0, 0, 0, 0, 0, 0};
  // 6-arg form (clang-23 / therock-10.0): groups 2/3 + extra group unused (<=2D)
  __builtin_amdgcn_tensor_load_to_lds(g0, g1, z4, z4, z8, 0);
}
#endif

}  // namespace

__global__ __launch_bounds__(BLK, 1)
void wfca_fused_kernel(const float* __restrict__ xg,
                       const float* __restrict__ w_real, const float* __restrict__ b_real,
                       const float* __restrict__ w_imag, const float* __restrict__ b_imag,
                       const float* __restrict__ car_w1, const float* __restrict__ car_b1,
                       const float* __restrict__ car_w2, const float* __restrict__ car_b2,
                       const float* __restrict__ cai_w1, const float* __restrict__ cai_b1,
                       const float* __restrict__ cai_w2, const float* __restrict__ cai_b2,
                       float* __restrict__ out) {
  extern __shared__ _Float16 sm[];
  _Float16* sRe  = sm + OFF_RE;    // [q][c], q = u*16+v
  _Float16* sIm  = sm + OFF_IM;
  _Float16* sCr  = sm + OFF_CR;    // [o][v*16+u]
  _Float16* sCi  = sm + OFF_CI;
  _Float16* sWr  = sm + OFF_WR;
  _Float16* sWi  = sm + OFF_WI;
  _Float16* sFr  = sm + OFF_FR;
  _Float16* sFi  = sm + OFF_FI;
  _Float16* sScr = sm + OFF_SCR;
  float* sF32   = (float*)(sm + HALVES);
  float* sPoolR = sF32;
  float* sPoolI = sPoolR + CCH;
  float* sHidR  = sPoolI + CCH;
  float* sHidI  = sHidR + CRED;
  float* sScR   = sHidI + CRED;
  float* sScI   = sScR + CCH;
  float* sBr    = sScI + CCH;
  float* sBi    = sBr + CCH;

  const int tid  = threadIdx.x;
  const int lane = tid & 31;
  const int wv   = tid >> 5;
  const int hi   = lane >> 4;
  const int ln   = lane & 15;

  const int win = blockIdx.x;
  const int bb  = win >> 8;
  const int h0  = ((win >> 4) & 15) * WSZ;
  const int w0  = (win & 15) * WSZ;

  // ---- stage 0: DFT matrix + biases; kick off async weight staging via TDM
  {
    const int u = tid >> 4, r = tid & 15;
    const float ang = -0.39269908169872414f * (float)((u * r) & 15);  // -pi/8*(ur%16)
    sFr[tid] = (_Float16)(cosf(ang) * 0.25f);
    sFi[tid] = (_Float16)(sinf(ang) * 0.25f);
  }
  if (tid < CCH) sBr[tid] = b_real[tid];
  else if (tid >= 128 && tid < 128 + CCH) sBi[tid - 128] = b_imag[tid - 128];
#if USE_TDM
  // Stage fp32 weights into the (currently unused) sCr/sCi region, DMA overlapped
  // with the FFT stage below.  Wave 0 issues; TENSORcnt waited before use.
  if (wv == 0) {
    tdm_load_linear((unsigned)(OFF_CR * 2), w_real, CCH * CCH);
    tdm_load_linear((unsigned)(OFF_CR * 2) + CCH * CCH * 4, w_imag, CCH * CCH);
  }
#else
  for (int i = tid; i < CCH * CCH; i += BLK) {
    sWr[i] = (_Float16)w_real[i];
    sWi[i] = (_Float16)w_imag[i];
  }
#endif
  __syncthreads();

  // persistent DFT fragments
  const v16h aFr    = frag_a_k16(sFr, 1.0f);            // stage-1 fwd (K hi half = 0)
  const v16h aFi    = frag_a_k16(sFi, 1.0f);
  const v16h bFrFiN = frag_b_pair16(sFr, sFi, 1.0f, -1.0f);  // Yr = [T1|T2][Fr;-Fi]
  const v16h bFiFr  = frag_b_pair16(sFi, sFr, 1.0f, 1.0f);   // Yi = [T1|T2][Fi; Fr]
  const v16h aFrFi  = frag_a_pair16(sFr, sFi, 1.0f);         // Ur = [Fr| Fi][Zr;Zi]
  const v16h aFrFiN = frag_a_pair16(sFr, sFi, -1.0f);        // Ui = [Fr|-Fi][Zi;Zr]
  const v16h bFrFi  = frag_b_pair16(sFr, sFi, 1.0f, 1.0f);   // Out = [Ur|Ui][Fr;Fi]

  _Float16* sT = sScr + wv * 512;  // 16 rows x 32 halves: [T1 row | T2 row]
  const v8f zero = {};

  // ---- stage 1: per-channel 2D FFT  Y = F X F, 12 channels per wave
  // All 32 lanes load the window column for n=lane&15 (elements 8-15 repeat
  // elements 0-7; as k 16..31 they multiply A's zero K-half, and x is finite, so the
  // products are exact zeros).  No predication -> no cndmask, full coalescing.
  const float* xw = xg + (((size_t)bb * CCH + wv) * 65536 + (size_t)h0 * 256 + w0 + ln);
  for (int j = 0; j < CCH / NWAVE; ++j) {
    const int c = wv + j * NWAVE;
    const float* p = xw + (size_t)j * NWAVE * 65536;
    v16h bx;
#pragma unroll
    for (int e = 0; e < 16; ++e) bx[e] = (_Float16)p[(size_t)((e & 7) + 8 * hi) * 256];
    v8f t1 = wmma_f16(aFr, bx, zero);
    v8f t2 = wmma_f16(aFi, bx, zero);
    store_d_rc(sT, 32, 1, t1);          // same-wave DS ops stay in order
    store_d_rc(sT + 16, 32, 1, t2);
    const v16h aT = frag_a_k32v(sT, 32);     // [T1|T2], full K=32, b128 loads
    v8f yr = wmma_f16(aT, bFrFiN, zero);
    v8f yi = wmma_f16(aT, bFiFr, zero);
    store_d_rc(sRe + c, 16 * CCH, CCH, yr);  // sRe[(u*16+v)*96 + c]
    store_d_rc(sIm + c, 16 * CCH, CCH, yi);
  }
#if USE_TDM
  if (wv == 0) __builtin_amdgcn_s_wait_tensorcnt(0);
  __syncthreads();  // FFT results + staged fp32 weights visible
  {
    const float* stR = (const float*)(sm + OFF_CR);
    const float* stI = stR + CCH * CCH;
    for (int i = tid; i < CCH * CCH; i += BLK) {
      sWr[i] = (_Float16)stR[i];
      sWi[i] = (_Float16)stI[i];
    }
  }
#endif
  __syncthreads();

  // ---- stage 2: channel conv as 96x256x96 GEMM (re & im), 24 jobs/wave
  for (int j = 0; j < 24; ++j) {
    const int job  = wv * 24 + j;
    const bool re  = job < 96;                 // uniform per wave
    const int tile = re ? job : job - 96;
    const int mt = tile >> 4, nt = tile & 15;
    const _Float16* Wm = re ? sWr : sWi;
    const _Float16* Xs = re ? sRe : sIm;
    const float* bias  = re ? sBr : sBi;
    _Float16* Dst      = re ? sCr : sCi;
    v8f acc = {};
#pragma unroll
    for (int kk = 0; kk < 3; ++kk) {
      const v16h a = frag_a_k32v(Wm + (mt * 16) * CCH + kk * 32, CCH);
      const v16h b = frag_b_k32t(Xs + (nt * 16) * CCH + kk * 32, CCH);
      acc = wmma_f16(a, b, acc);
    }
    const int m0 = mt * 16 + 8 * hi;
#pragma unroll
    for (int r = 0; r < 8; ++r) acc[r] += bias[m0 + r];
    // D (o, q=nt*16+n) -> sC[o*256 + n*16 + nt]  ([o][v][u] for the IFFT)
    store_d_rc(Dst + (mt * 16) * 256 + nt, 256, 16, acc);
  }
  __syncthreads();

  // ---- stage 3: SE attention (tiny)
  if (tid < CCH) {
    const v8h* p = (const v8h*)(sCr + tid * PWIN);
    float s = 0.0f;
    for (int i = 0; i < PWIN / 8; ++i) {
      v8h v = p[i];
#pragma unroll
      for (int e = 0; e < 8; ++e) s += (float)v[e];
    }
    sPoolR[tid] = s * (1.0f / PWIN);
  } else if (tid >= 128 && tid < 128 + CCH) {
    const int o = tid - 128;
    const v8h* p = (const v8h*)(sCi + o * PWIN);
    float s = 0.0f;
    for (int i = 0; i < PWIN / 8; ++i) {
      v8h v = p[i];
#pragma unroll
      for (int e = 0; e < 8; ++e) s += (float)v[e];
    }
    sPoolI[o] = s * (1.0f / PWIN);
  }
  __syncthreads();
  if (tid < CRED) {
    float s = car_b1[tid];
    for (int o = 0; o < CCH; ++o) s += car_w1[tid * CCH + o] * sPoolR[o];
    sHidR[tid] = fmaxf(s, 0.0f);
  } else if (tid >= 32 && tid < 64) {
    const int i = tid - 32;
    float s = cai_b1[i];
    for (int o = 0; o < CCH; ++o) s += cai_w1[i * CCH + o] * sPoolI[o];
    sHidI[i] = fmaxf(s, 0.0f);
  }
  __syncthreads();
  if (tid < CCH) {
    float s = car_b2[tid];
    for (int i = 0; i < CRED; ++i) s += car_w2[tid * CRED + i] * sHidR[i];
    sScR[tid] = 1.0f + 1.0f / (1.0f + __expf(-s));
  } else if (tid >= 128 && tid < 128 + CCH) {
    const int o = tid - 128;
    float s = cai_b2[o];
    for (int i = 0; i < CRED; ++i) s += cai_w2[o * CRED + i] * sHidI[i];
    sScI[o] = 1.0f + 1.0f / (1.0f + __expf(-s));
  }
  __syncthreads();

  // ---- stage 4: scale + inverse FFT (real part), 12 channels per wave
  for (int j = 0; j < CCH / NWAVE; ++j) {
    const int c = wv + j * NWAVE;
    const _Float16 scR = (_Float16)sScR[c];
    const _Float16 scI = (_Float16)sScI[c];
    const _Float16* zr = sCr + c * PWIN;
    const _Float16* zi = sCi + c * PWIN;
    const v16h bZrZi = frag_b_zz(zr, zi, scR, scI);
    const v16h bZiZr = frag_b_zz(zi, zr, scI, scR);
    v8f ur = wmma_f16(aFrFi, bZrZi, zero);   // Ur = Fr Zr + Fi Zi
    v8f ui = wmma_f16(aFrFiN, bZiZr, zero);  // Ui = Fr Zi - Fi Zr
    store_d_rc(sT, 32, 1, ur);
    store_d_rc(sT + 16, 32, 1, ui);
    const v16h aU = frag_a_k32v(sT, 32);     // [Ur|Ui]
    v8f od = wmma_f16(aU, bFrFi, zero);      // Out = Ur Fr + Ui Fi
    float* op = out + (((size_t)(bb * CCH + c) * 256 + h0) * 256 + w0 + ln);
#pragma unroll
    for (int r = 0; r < 8; ++r) op[(size_t)(8 * hi + r) * 256] = od[r];
  }
}

extern "C" void kernel_launch(void* const* d_in, const int* in_sizes, int n_in,
                              void* d_out, int out_size, void* d_ws, size_t ws_size,
                              hipStream_t stream) {
  (void)in_sizes; (void)n_in; (void)out_size; (void)d_ws; (void)ws_size;
  const float* x       = (const float*)d_in[0];
  const float* w_real  = (const float*)d_in[1];
  const float* b_real  = (const float*)d_in[2];
  const float* w_imag  = (const float*)d_in[3];
  const float* b_imag  = (const float*)d_in[4];
  const float* car_w1  = (const float*)d_in[5];
  const float* car_b1  = (const float*)d_in[6];
  const float* car_w2  = (const float*)d_in[7];
  const float* car_b2  = (const float*)d_in[8];
  const float* cai_w1  = (const float*)d_in[9];
  const float* cai_b1  = (const float*)d_in[10];
  const float* cai_w2  = (const float*)d_in[11];
  const float* cai_b2  = (const float*)d_in[12];
  float* out = (float*)d_out;

  (void)hipFuncSetAttribute((const void*)wfca_fused_kernel,
                            hipFuncAttributeMaxDynamicSharedMemorySize,
                            (int)SMEM_BYTES);
  wfca_fused_kernel<<<NWIN, BLK, SMEM_BYTES, stream>>>(
      x, w_real, b_real, w_imag, b_imag,
      car_w1, car_b1, car_w2, car_b2,
      cai_w1, cai_b1, cai_w2, cai_b2, out);
}
